// ViT_69965017252597
// MI455X (gfx1250) — compile-verified, matrix operand order checked by reading, not run on hardware
//
#include <hip/hip_runtime.h>
#include <hip/hip_bf16.h>

// ---------------------------------------------------------------------------
// CDNA5 (gfx1250) ViT encoder-decoder forward.
// All matmuls run through v_wmma_f32_16x16x32_f16 (f32 accumulate), with
// fp32 -> f16 conversion performed while staging tiles into LDS.
// Fast path (aligned, K%32==0): 128x128 block, 8 waves of 64x32 each,
// global_load_b128 staging pipelined against WMMA. Scalar path (BT) keeps
// 128x64 blocks for the P@V attention GEMMs.
// ---------------------------------------------------------------------------

typedef __attribute__((ext_vector_type(16))) _Float16 v16h;
typedef __attribute__((ext_vector_type(8)))  _Float16 v8h;
typedef __attribute__((ext_vector_type(4)))  _Float16 v4h;
typedef __attribute__((ext_vector_type(8)))  float    v8f;

#define BM 128
#define BK 32
#define LDSP 40   // padded LDS row stride in halves (80 bytes, 16B/8B aligned)

enum { EPI_NONE = 0, EPI_BIAS = 1, EPI_BIAS_GELU = 2, EPI_SCALE = 3 };

union AFrag { v16h v; v8h h[2]; };

// D[M,N] = epi( A[M,K] @ B'[N,K]^T + bias )
//   !BT: B' row-major [N,K]  (weights; AL fast path requires K%32==0 and
//        16B-aligned rows, true for every !BT call in this model)
//    BT: B memory row-major [K,N], transposed during LDS staging (P@V)
template <int EPI, bool BT, bool AL>
__global__ __launch_bounds__(256)
void gemm_wmma_kernel(const float* __restrict__ A, const float* __restrict__ Bm,
                      const float* __restrict__ bias, float* __restrict__ C,
                      int M, int N, int K, int lda, int ldb, int ldc,
                      long sAb, long sAh, long sBb, long sBh, long sCb, long sCh,
                      int Hdim, float scale)
{
    constexpr int BN = AL ? 128 : 64;   // block N-tile
    constexpr int MT = AL ? 4 : 2;      // 16-row WMMA tiles per wave (M)
    constexpr int NT = 2;               // 16-col WMMA tiles per wave (N)

    __shared__ __attribute__((aligned(16))) _Float16 As[BM * LDSP];
    __shared__ __attribute__((aligned(16))) _Float16 Bs[BN * LDSP];

    const int z  = blockIdx.z;
    const int bb = z / Hdim, hh = z % Hdim;
    A  += (long)bb * sAb + (long)hh * sAh;
    Bm += (long)bb * sBb + (long)hh * sBh;
    C  += (long)bb * sCb + (long)hh * sCh;

    const int m0 = blockIdx.x * BM, n0 = blockIdx.y * BN;
    const int tid  = threadIdx.x;
    const int lane = tid & 31, wave = tid >> 5;
    // wave layout: AL -> 2(M) x 4(N), each wave 64x32 ; !AL -> 4x2, 32x32
    const int wm = AL ? (wave & 1) : (wave & 3);
    const int wn = AL ? (wave >> 1) : (wave >> 2);
    const int laneHi = (lane >> 4) & 1;
    const int lane15 = lane & 15;

    v8f acc[MT][NT] = {};

    // register staging buffers (double-buffer vs. LDS)
    float4 ra[4], rb[AL ? 4 : 1];       // fast path (B unused slot when !AL)
    float  sa[16], sb[8];               // scalar path

    // ---- load K-tile at kk into registers (branch-free, clamped) ----
    auto load_regs = [&](int kk) {
        if constexpr (AL) {
#pragma unroll
            for (int i = 0; i < 4; ++i) {              // A: 128x32 = 1024 float4
                int linear = i * 256 + tid;
                int r = linear >> 3, c4 = linear & 7;
                int gm = m0 + r; gm = gm < M ? gm : M - 1;
                ra[i] = *(const float4*)(A + (long)gm * lda + kk + c4 * 4);
            }
#pragma unroll
            for (int i = 0; i < 4; ++i) {              // B: 128x32 = 1024 float4
                int linear = i * 256 + tid;
                int r = linear >> 3, c4 = linear & 7;
                int gn = n0 + r; gn = gn < N ? gn : N - 1;
                rb[i] = *(const float4*)(Bm + (long)gn * ldb + kk + c4 * 4);
            }
        } else {
#pragma unroll
            for (int i = 0; i < 16; ++i) {
                int linear = i * 256 + tid;
                int r = linear >> 5, c = linear & 31;
                int gm = m0 + r; gm = gm < M ? gm : M - 1;
                int gk = kk + c;
                float v = A[(long)gm * lda + (gk < K ? gk : K - 1)];
                sa[i] = gk < K ? v : 0.0f;
            }
#pragma unroll
            for (int i = 0; i < 8; ++i) {
                int linear = i * 256 + tid;
                if (!BT) {
                    int r = linear >> 5, c = linear & 31;   // r = n, c = k
                    int gn = n0 + r; gn = gn < N ? gn : N - 1;
                    int gk = kk + c;
                    float v = Bm[(long)gn * ldb + (gk < K ? gk : K - 1)];
                    sb[i] = gk < K ? v : 0.0f;
                } else {
                    int c = linear >> 6, n = linear & 63;   // c = k, n = col
                    int gn = n0 + n; gn = gn < N ? gn : N - 1;
                    int gk = kk + c;
                    float v = Bm[(long)(gk < K ? gk : K - 1) * ldb + gn];
                    sb[i] = gk < K ? v : 0.0f;
                }
            }
        }
    };

    // ---- store register tile into LDS as f16 ----
    auto store_lds = [&]() {
        if constexpr (AL) {
#pragma unroll
            for (int i = 0; i < 4; ++i) {
                int linear = i * 256 + tid;
                int r = linear >> 3, c4 = linear & 7;
                v4h h;
                h[0] = (_Float16)ra[i].x; h[1] = (_Float16)ra[i].y;
                h[2] = (_Float16)ra[i].z; h[3] = (_Float16)ra[i].w;
                *(v4h*)&As[r * LDSP + c4 * 4] = h;
            }
#pragma unroll
            for (int i = 0; i < 4; ++i) {
                int linear = i * 256 + tid;
                int r = linear >> 3, c4 = linear & 7;
                v4h h;
                h[0] = (_Float16)rb[i].x; h[1] = (_Float16)rb[i].y;
                h[2] = (_Float16)rb[i].z; h[3] = (_Float16)rb[i].w;
                *(v4h*)&Bs[r * LDSP + c4 * 4] = h;
            }
        } else {
#pragma unroll
            for (int i = 0; i < 16; ++i) {
                int linear = i * 256 + tid;
                int r = linear >> 5, c = linear & 31;
                As[r * LDSP + c] = (_Float16)sa[i];
            }
#pragma unroll
            for (int i = 0; i < 8; ++i) {
                int linear = i * 256 + tid;
                if (!BT) {
                    int r = linear >> 5, c = linear & 31;
                    Bs[r * LDSP + c] = (_Float16)sb[i];
                } else {
                    int c = linear >> 6, n = linear & 63;
                    Bs[n * LDSP + c] = (_Float16)sb[i];
                }
            }
        }
    };

    load_regs(0);                                   // prologue
    for (int kk = 0; kk < K; kk += BK) {
        store_lds();
        __syncthreads();

        if (kk + BK < K) load_regs(kk + BK);        // overlap with WMMA below

        // ---- fragment loads (two 16B ds loads per operand) ----
        const int kb = laneHi ? 8 : 0;   // halves: lanes16-31 take K+8 block
        AFrag a[MT], b[NT];
#pragma unroll
        for (int t = 0; t < MT; ++t) {
            int ar = wm * (MT * 16) + t * 16 + lane15;
            a[t].h[0] = *(const v8h*)&As[ar * LDSP + kb];
            a[t].h[1] = *(const v8h*)&As[ar * LDSP + kb + 16];
        }
#pragma unroll
        for (int t = 0; t < NT; ++t) {
            int br = wn * 32 + t * 16 + lane15;
            b[t].h[0] = *(const v8h*)&Bs[br * LDSP + kb];
            b[t].h[1] = *(const v8h*)&Bs[br * LDSP + kb + 16];
        }
#pragma unroll
        for (int ti = 0; ti < MT; ++ti)
#pragma unroll
            for (int tj = 0; tj < NT; ++tj)
                acc[ti][tj] = __builtin_amdgcn_wmma_f32_16x16x32_f16(
                    false, a[ti].v, false, b[tj].v,
                    (short)0, acc[ti][tj], false, false);
        __syncthreads();
    }

    // ---- epilogue: C/D layout = VGPR v -> M = v + laneHi*8, N = lane&15 ----
#pragma unroll
    for (int ti = 0; ti < MT; ++ti)
#pragma unroll
        for (int tj = 0; tj < NT; ++tj) {
            int col = n0 + wn * 32 + tj * 16 + lane15;
            if (col >= N) continue;
            float bv = 0.0f;
            if (EPI == EPI_BIAS || EPI == EPI_BIAS_GELU) bv = bias[col];
#pragma unroll
            for (int v = 0; v < 8; ++v) {
                int row = m0 + wm * (MT * 16) + ti * 16 + v + laneHi * 8;
                if (row >= M) continue;
                float x = acc[ti][tj][v];
                if (EPI == EPI_BIAS) x += bv;
                else if (EPI == EPI_BIAS_GELU) {
                    x += bv;
                    x = 0.5f * x * (1.0f + erff(x * 0.70710678118654752f));
                } else if (EPI == EPI_SCALE) x *= scale;
                C[(long)row * ldc + col] = x;
            }
        }
}

// row-wise softmax, one wave32 per row; optional causal mask (col > row%Mrow)
__global__ __launch_bounds__(256)
void softmax_rows_kernel(float* __restrict__ S, long rows, int cols, int ld,
                         int Mrow, int causal)
{
    long wid = ((long)blockIdx.x * blockDim.x + threadIdx.x) >> 5;
    int lane = threadIdx.x & 31;
    if (wid >= rows) return;
    float* row = S + wid * (long)ld;
    int i = (int)(wid % Mrow);
    float mx = -3.4e38f;
    for (int j = lane; j < cols; j += 32) {
        float v = row[j];
        if (causal && j > i) v = -1e9f;
        mx = fmaxf(mx, v);
    }
    for (int o = 16; o >= 1; o >>= 1) mx = fmaxf(mx, __shfl_xor(mx, o, 32));
    float sum = 0.0f;
    for (int j = lane; j < cols; j += 32) {
        float v = row[j];
        if (causal && j > i) v = -1e9f;
        v = __expf(v - mx);
        row[j] = v;
        sum += v;
    }
    for (int o = 16; o >= 1; o >>= 1) sum += __shfl_xor(sum, o, 32);
    float inv = 1.0f / sum;
    for (int j = lane; j < cols; j += 32) row[j] *= inv;
}

// Out[r,:] = LayerNorm(X[r,:] (+ R[r,:])) * g + b   for C = 768, wave per row
__global__ __launch_bounds__(256)
void ln768_kernel(const float* __restrict__ X, const float* __restrict__ R,
                  const float* __restrict__ g, const float* __restrict__ b,
                  float* __restrict__ Out, long rows)
{
    long wid = ((long)blockIdx.x * blockDim.x + threadIdx.x) >> 5;
    int lane = threadIdx.x & 31;
    if (wid >= rows) return;
    const float* x = X + wid * 768;
    const float* r = R ? R + wid * 768 : nullptr;
    float v[24];
    float s = 0.0f;
#pragma unroll
    for (int i = 0; i < 24; ++i) {
        float t = x[lane + i * 32];
        if (r) t += r[lane + i * 32];
        v[i] = t; s += t;
    }
    for (int o = 16; o >= 1; o >>= 1) s += __shfl_xor(s, o, 32);
    float mean = s * (1.0f / 768.0f);
    float var = 0.0f;
#pragma unroll
    for (int i = 0; i < 24; ++i) { float d = v[i] - mean; var += d * d; }
    for (int o = 16; o >= 1; o >>= 1) var += __shfl_xor(var, o, 32);
    float inv = rsqrtf(var * (1.0f / 768.0f) + 1e-5f);
    float* out = Out + wid * 768;
#pragma unroll
    for (int i = 0; i < 24; ++i) {
        int c = lane + i * 32;
        out[c] = (v[i] - mean) * inv * g[c] + b[c];
    }
}

// hdn[b, n, c] = (n==0 ? cls[c] : pch[b, n-1, c]) + pos[n, c]
__global__ void assemble_hdn_kernel(const float* __restrict__ pch,
                                    const float* __restrict__ cls,
                                    const float* __restrict__ pos,
                                    float* __restrict__ hdn, long total)
{
    long idx = (long)blockIdx.x * blockDim.x + threadIdx.x;
    if (idx >= total) return;
    int c = (int)(idx % 768);
    long t = idx / 768;
    int n = (int)(t % 197);
    long bb = t / 197;
    float v = (n == 0) ? cls[c] : pch[(bb * 196 + (n - 1)) * 768 + c];
    hdn[idx] = v + pos[(long)n * 768 + c];
}

// tgt[b, t, c] = emb_out[y[b,t], c] + pos_out[t, c]
__global__ void build_tgt_kernel(const int* __restrict__ y,
                                 const float* __restrict__ emb_out,
                                 const float* __restrict__ pos_out,
                                 float* __restrict__ tgt, long total, int T)
{
    long idx = (long)blockIdx.x * blockDim.x + threadIdx.x;
    if (idx >= total) return;
    int c = (int)(idx % 768);
    long t = idx / 768;
    int tt = (int)(t % T);
    long bb = t / T;
    int tok = y[bb * T + tt];
    tgt[idx] = emb_out[(long)tok * 768 + c] + pos_out[(long)tt * 768 + c];
}

// ---------------------------------------------------------------------------
// Host-side orchestration
// ---------------------------------------------------------------------------

static void launch_gemm(hipStream_t st, int epi, bool bt,
                        const float* A, const float* B, const float* bias, float* C,
                        int M, int N, int K, int lda, int ldb, int ldc,
                        long sAb, long sAh, long sBb, long sBh, long sCb, long sCh,
                        int Hdim, int batch, float scale)
{
    int bn = bt ? 64 : 128;
    dim3 g((M + BM - 1) / BM, (N + bn - 1) / bn, batch), blk(256);
#define GL(E, T_, AL_) gemm_wmma_kernel<E, T_, AL_><<<g, blk, 0, st>>>(A, B, bias, C, \
        M, N, K, lda, ldb, ldc, sAb, sAh, sBb, sBh, sCb, sCh, Hdim, scale)
    if (!bt) {   // aligned fast path: K%32==0 and 16B-aligned rows (all !BT calls)
        switch (epi) { case EPI_NONE: GL(EPI_NONE,false,true); break;
                       case EPI_BIAS: GL(EPI_BIAS,false,true); break;
                       case EPI_BIAS_GELU: GL(EPI_BIAS_GELU,false,true); break;
                       default: GL(EPI_SCALE,false,true); }
    } else {
        switch (epi) { case EPI_NONE: GL(EPI_NONE,true,false); break;
                       case EPI_BIAS: GL(EPI_BIAS,true,false); break;
                       case EPI_BIAS_GELU: GL(EPI_BIAS_GELU,true,false); break;
                       default: GL(EPI_SCALE,true,false); }
    }
#undef GL
}

static inline void launch_softmax(hipStream_t st, float* S, long rows, int cols,
                                  int Mrow, int causal)
{
    long blocks = (rows + 7) / 8;
    softmax_rows_kernel<<<dim3((unsigned)blocks), dim3(256), 0, st>>>(
        S, rows, cols, cols, Mrow, causal);
}

static inline void launch_ln(hipStream_t st, const float* X, const float* R,
                             const float* g, const float* b, float* Out, long rows)
{
    long blocks = (rows + 7) / 8;
    ln768_kernel<<<dim3((unsigned)blocks), dim3(256), 0, st>>>(X, R, g, b, Out, rows);
}

extern "C" void kernel_launch(void* const* d_in, const int* in_sizes, int n_in,
                              void* d_out, int out_size, void* d_ws, size_t ws_size,
                              hipStream_t stream)
{
    (void)in_sizes; (void)n_in; (void)out_size; (void)ws_size;
    constexpr int Bsz = 64, Nn = 197, NPp = 196, Cc = 768, Hh = 12, Tt = 7, Ll = 12, NCc = 10;
    constexpr long TOK_E = (long)Bsz * Nn;   // 12608
    constexpr long TOK_P = (long)Bsz * NPp;  // 12544
    constexpr long TOK_D = (long)Bsz * Tt;   // 448
    constexpr long SZ    = TOK_E * Cc;       // encoder activation size

    auto P = [&](int i) -> const float* { return (const float*)d_in[i]; };
    const float* xin = P(0);
    const int*   yin = (const int*)d_in[1];
    // params flat order: emb_w(2) emb_b(3) cls(4) pos(5) emb_out(6) pos_out(7)
    // enc layers @8 (16 each), dec layers @200 (26 each), fin @512..515

    float* W   = (float*)d_ws;
    float* hdn = W;
    float* bq  = W + 1 * SZ;
    float* bk  = W + 2 * SZ;
    float* bv  = W + 3 * SZ;
    float* bo  = W + 4 * SZ;
    float* by  = W + 5 * SZ;
    float* sc  = W + 6 * SZ;                               // [B,H,197,197]
    float* tgt = sc + (long)Bsz * Hh * Nn * Nn;
    float* tq  = tgt + TOK_D * Cc;
    float* tk  = tq  + TOK_D * Cc;
    float* tv  = tk  + TOK_D * Cc;
    float* to_ = tv  + TOK_D * Cc;
    float* ty  = to_ + TOK_D * Cc;
    float* ds  = ty  + TOK_D * Cc;                         // dec scores
    float* lnf = ds  + (long)Bsz * Hh * Tt * Nn;

    const float iscl = 0.125f;  // 1/sqrt(64)

    // ---- patch embed: pch = x @ emb_w^T + emb_b  (into bq as scratch) ----
    launch_gemm(stream, EPI_BIAS, false, xin, P(2), P(3), bq,
                (int)TOK_P, Cc, Cc, Cc, Cc, Cc, 0,0,0,0,0,0, 1, 1, 1.f);
    {
        long total = TOK_E * Cc;
        assemble_hdn_kernel<<<dim3((unsigned)((total + 255) / 256)), dim3(256), 0, stream>>>(
            bq, P(4), P(5), hdn, total);
    }

    // ---- encoder ----
    for (int l = 0; l < Ll; ++l) {
        int bi = 8 + 16 * l;
        launch_gemm(stream, EPI_BIAS, false, hdn, P(bi+0), P(bi+1), bq,
                    (int)TOK_E, Cc, Cc, Cc, Cc, Cc, 0,0,0,0,0,0, 1, 1, 1.f);
        launch_gemm(stream, EPI_BIAS, false, hdn, P(bi+2), P(bi+3), bk,
                    (int)TOK_E, Cc, Cc, Cc, Cc, Cc, 0,0,0,0,0,0, 1, 1, 1.f);
        launch_gemm(stream, EPI_BIAS, false, hdn, P(bi+4), P(bi+5), bv,
                    (int)TOK_E, Cc, Cc, Cc, Cc, Cc, 0,0,0,0,0,0, 1, 1, 1.f);
        // scores = Q K^T / sqrt(hd) : batched over (b,h)
        launch_gemm(stream, EPI_SCALE, false, bq, bk, nullptr, sc,
                    Nn, Nn, 64, Cc, Cc, Nn,
                    (long)Nn*Cc, 64, (long)Nn*Cc, 64,
                    (long)Hh*Nn*Nn, (long)Nn*Nn, Hh, Bsz*Hh, iscl);
        launch_softmax(stream, sc, (long)Bsz*Hh*Nn, Nn, Nn, 0);
        // O = P @ V  (V memory is [K=197, N=64], transpose-staged)
        launch_gemm(stream, EPI_NONE, true, sc, bv, nullptr, bo,
                    Nn, 64, Nn, Nn, Cc, Cc,
                    (long)Hh*Nn*Nn, (long)Nn*Nn, (long)Nn*Cc, 64,
                    (long)Nn*Cc, 64, Hh, Bsz*Hh, 1.f);
        launch_gemm(stream, EPI_BIAS, false, bo, P(bi+6), P(bi+7), by,
                    (int)TOK_E, Cc, Cc, Cc, Cc, Cc, 0,0,0,0,0,0, 1, 1, 1.f);
        launch_ln(stream, hdn, by, P(bi+8), P(bi+9), hdn, TOK_E);
        // FFN
        launch_gemm(stream, EPI_BIAS_GELU, false, hdn, P(bi+10), P(bi+11), by,
                    (int)TOK_E, Cc, Cc, Cc, Cc, Cc, 0,0,0,0,0,0, 1, 1, 1.f);
        launch_gemm(stream, EPI_BIAS, false, by, P(bi+12), P(bi+13), bo,
                    (int)TOK_E, Cc, Cc, Cc, Cc, Cc, 0,0,0,0,0,0, 1, 1, 1.f);
        launch_ln(stream, hdn, bo, P(bi+14), P(bi+15), hdn, TOK_E);
    }

    // ---- decoder input ----
    {
        long total = TOK_D * Cc;
        build_tgt_kernel<<<dim3((unsigned)((total + 255) / 256)), dim3(256), 0, stream>>>(
            yin, P(6), P(7), tgt, total, Tt);
    }

    // ---- decoder ----
    for (int l = 0; l < Ll; ++l) {
        int bi = 200 + 26 * l;
        // self-attention (causal)
        launch_gemm(stream, EPI_BIAS, false, tgt, P(bi+0), P(bi+1), tq,
                    (int)TOK_D, Cc, Cc, Cc, Cc, Cc, 0,0,0,0,0,0, 1, 1, 1.f);
        launch_gemm(stream, EPI_BIAS, false, tgt, P(bi+2), P(bi+3), tk,
                    (int)TOK_D, Cc, Cc, Cc, Cc, Cc, 0,0,0,0,0,0, 1, 1, 1.f);
        launch_gemm(stream, EPI_BIAS, false, tgt, P(bi+4), P(bi+5), tv,
                    (int)TOK_D, Cc, Cc, Cc, Cc, Cc, 0,0,0,0,0,0, 1, 1, 1.f);
        launch_gemm(stream, EPI_SCALE, false, tq, tk, nullptr, ds,
                    Tt, Tt, 64, Cc, Cc, Tt,
                    (long)Tt*Cc, 64, (long)Tt*Cc, 64,
                    (long)Hh*Tt*Tt, (long)Tt*Tt, Hh, Bsz*Hh, iscl);
        launch_softmax(stream, ds, (long)Bsz*Hh*Tt, Tt, Tt, 1);
        launch_gemm(stream, EPI_NONE, true, ds, tv, nullptr, to_,
                    Tt, 64, Tt, Tt, Cc, Cc,
                    (long)Hh*Tt*Tt, (long)Tt*Tt, (long)Tt*Cc, 64,
                    (long)Tt*Cc, 64, Hh, Bsz*Hh, 1.f);
        launch_gemm(stream, EPI_BIAS, false, to_, P(bi+6), P(bi+7), ty,
                    (int)TOK_D, Cc, Cc, Cc, Cc, Cc, 0,0,0,0,0,0, 1, 1, 1.f);
        launch_ln(stream, tgt, ty, P(bi+16), P(bi+17), tgt, TOK_D);
        // cross-attention (K/V from encoder output)
        launch_gemm(stream, EPI_BIAS, false, tgt, P(bi+8), P(bi+9), tq,
                    (int)TOK_D, Cc, Cc, Cc, Cc, Cc, 0,0,0,0,0,0, 1, 1, 1.f);
        launch_gemm(stream, EPI_BIAS, false, hdn, P(bi+10), P(bi+11), bk,
                    (int)TOK_E, Cc, Cc, Cc, Cc, Cc, 0,0,0,0,0,0, 1, 1, 1.f);
        launch_gemm(stream, EPI_BIAS, false, hdn, P(bi+12), P(bi+13), bv,
                    (int)TOK_E, Cc, Cc, Cc, Cc, Cc, 0,0,0,0,0,0, 1, 1, 1.f);
        launch_gemm(stream, EPI_SCALE, false, tq, bk, nullptr, ds,
                    Tt, Nn, 64, Cc, Cc, Nn,
                    (long)Tt*Cc, 64, (long)Nn*Cc, 64,
                    (long)Hh*Tt*Nn, (long)Tt*Nn, Hh, Bsz*Hh, iscl);
        launch_softmax(stream, ds, (long)Bsz*Hh*Tt, Nn, Tt, 0);
        launch_gemm(stream, EPI_NONE, true, ds, bv, nullptr, to_,
                    Tt, 64, Nn, Nn, Cc, Cc,
                    (long)Hh*Tt*Nn, (long)Tt*Nn, (long)Nn*Cc, 64,
                    (long)Tt*Cc, 64, Hh, Bsz*Hh, 1.f);
        launch_gemm(stream, EPI_BIAS, false, to_, P(bi+14), P(bi+15), ty,
                    (int)TOK_D, Cc, Cc, Cc, Cc, Cc, 0,0,0,0,0,0, 1, 1, 1.f);
        launch_ln(stream, tgt, ty, P(bi+18), P(bi+19), tgt, TOK_D);
        // FFN
        launch_gemm(stream, EPI_BIAS_GELU, false, tgt, P(bi+20), P(bi+21), ty,
                    (int)TOK_D, Cc, Cc, Cc, Cc, Cc, 0,0,0,0,0,0, 1, 1, 1.f);
        launch_gemm(stream, EPI_BIAS, false, ty, P(bi+22), P(bi+23), to_,
                    (int)TOK_D, Cc, Cc, Cc, Cc, Cc, 0,0,0,0,0,0, 1, 1, 1.f);
        launch_ln(stream, tgt, to_, P(bi+24), P(bi+25), tgt, TOK_D);
    }

    // ---- final LN -> classifier -> softmax into d_out ----
    launch_ln(stream, tgt, nullptr, P(512), P(513), lnf, TOK_D);
    launch_gemm(stream, EPI_BIAS, false, lnf, P(514), P(515), (float*)d_out,
                (int)TOK_D, NCc, Cc, Cc, Cc, NCc, 0,0,0,0,0,0, 1, 1, 1.f);
    launch_softmax(stream, (float*)d_out, TOK_D, NCc, 1, 0);
}
